// RqVae_40467181863115
// MI455X (gfx1250) — matrix operand order repaired and checked by reading, not compiled, not run on hardware
//
#include <hip/hip_runtime.h>
#include <hip/hip_bf16.h>

// ---------------------------------------------------------------------------
// RQ-VAE forward for MI455X (gfx1250, wave32, WMMA).
//   GEMMs + VQ similarity on v_wmma_f32_16x16x32_bf16 (fp32 accumulate).
//   Argmax fused into the similarity GEMM (sim matrix never hits HBM).
//   Global->LDS uses GLOBAL_LOAD_ASYNC_TO_LDS_B128 (ASYNCcnt) + LDS double
//   buffering when the toolchain exposes the builtin; sync fallback otherwise.
// ---------------------------------------------------------------------------

#define NROWS 32768
#define IN_DIM 768
#define HID 1024
#define EMB 256
#define KCODES 2048
#define KCLUST 50
#define KCLUST_PAD 128

typedef __attribute__((ext_vector_type(16))) __bf16 bf16x16;
typedef __attribute__((ext_vector_type(8)))  float  f32x8;
typedef __attribute__((ext_vector_type(4)))  int    v4i;

#define KSTEP 32
#define LSTR  40    // 32-wide K-slice tile row stride (80B rows, conflict-free)
#define AFSTR 264   // full 256-wide A tile row stride (528B rows, conflict-free)

#if defined(__AMDGCN__) && __has_builtin(__builtin_amdgcn_global_load_async_to_lds_b128)
#define USE_ASYNC 1
#else
#define USE_ASYNC 0
#endif

#if USE_ASYNC
#if __has_builtin(__builtin_amdgcn_s_wait_asynccnt)
#define WAIT_ASYNC(n) __builtin_amdgcn_s_wait_asynccnt(n)
#else
#define WAIT_ASYNC(n) asm volatile("s_wait_asynccnt %0" :: "n"(n) : "memory")
#endif
#endif

struct U128x2 { uint4 a, b; };

// --- sync path: global(128x32 bf16 tile) -> regs -> LDS --------------------
__device__ __forceinline__ void lds_load_tile(const __bf16* __restrict__ g, int ldg,
                                              __bf16* __restrict__ s,
                                              int row0, int k0, int tid) {
#pragma unroll
  for (int it = 0; it < 2; ++it) {
    int id  = tid + it * 256;      // 0..511 -> 512 uint4 = 128 rows * 64B
    int r   = id >> 2;
    int seg = id & 3;
    uint4 v = *(const uint4*)(g + (size_t)(row0 + r) * ldg + k0 + seg * 8);
    *(uint4*)(s + r * LSTR + seg * 8) = v;
  }
}

#if USE_ASYNC
// --- async path: GLOBAL_LOAD_ASYNC_TO_LDS_B128, per-lane LDS address -------
// builtin signature (from clang diagnostic): (v4i AS1*, v4i AS3*, Imm, Imm)
__device__ __forceinline__ void async_b128(const __bf16* g, __bf16* s) {
  __builtin_amdgcn_global_load_async_to_lds_b128(
      (__attribute__((address_space(1))) v4i*)g,
      (__attribute__((address_space(3))) v4i*)s, 0, 0);
}
// 128 rows x 32 cols bf16 (one K-slice): 2 async b128 per thread
__device__ __forceinline__ void async_tile32(const __bf16* __restrict__ g, int ldg,
                                             __bf16* __restrict__ s,
                                             int row0, int k0, int tid) {
#pragma unroll
  for (int it = 0; it < 2; ++it) {
    int id = tid + it * 256, r = id >> 2, seg = id & 3;
    async_b128(g + (size_t)(row0 + r) * ldg + k0 + seg * 8, s + r * LSTR + seg * 8);
  }
}
// 128 rows x 256 cols bf16 (full A tile): 16 async b128 per thread
__device__ __forceinline__ void async_tile_fullA(const __bf16* __restrict__ g,
                                                 __bf16* __restrict__ s,
                                                 int row0, int tid) {
#pragma unroll
  for (int it = 0; it < 16; ++it) {
    int id = tid + it * 256;       // 0..4095 uint4
    int r = id >> 5, seg = id & 31;
    async_b128(g + (size_t)(row0 + r) * EMB + seg * 8, s + r * AFSTR + seg * 8);
  }
}
#endif

// --- A fragment 16x32 bf16 per ISA 7.12.2: lane m=l%16; K half (l/16)*8 ----
template<int STR>
__device__ __forceinline__ bf16x16 frag_a(const __bf16* __restrict__ s, int lane) {
  int m  = lane & 15;
  int kg = (lane >> 4) << 3;                 // 0 or 8
  const __bf16* p = s + m * STR + kg;
  U128x2 t;
  t.a = *(const uint4*)p;                    // K = kg .. kg+7
  t.b = *(const uint4*)(p + 16);             // K = kg+16 .. kg+23
  return __builtin_bit_cast(bf16x16, t);
}

// --- B fragment 32x16 bf16: lane n=l%16; K half (l/16)*8 (same as A) -------
template<int STR>
__device__ __forceinline__ bf16x16 frag_b(const __bf16* __restrict__ s, int lane) {
  int n  = lane & 15;
  int kg = (lane >> 4) << 3;                 // 0 or 8
  const __bf16* p = s + n * STR + kg;
  U128x2 t;
  t.a = *(const uint4*)p;
  t.b = *(const uint4*)(p + 16);
  return __builtin_bit_cast(bf16x16, t);
}

// ---------------------------------------------------------------------------
// NT GEMM: C[M,N] = act(A[M,Kd] * Bt[N,Kd]^T + bias). Block 256 thr = 8 waves.
// Tile 128x128; wave w owns rows [w*16, w*16+16) x all 128 cols (8 wmma tiles).
// ---------------------------------------------------------------------------
template<int RELU, int OUT_BF16>
__global__ __launch_bounds__(256)
void wmma_gemm_nt(const __bf16* __restrict__ A, int lda,
                  const __bf16* __restrict__ Bt, int ldb,
                  const float* __restrict__ bias,
                  void* __restrict__ Cout, int ldc, int Kd) {
  __shared__ __bf16 As[2][128 * LSTR];
  __shared__ __bf16 Bs[2][128 * LSTR];
  const int tid = threadIdx.x, lane = tid & 31, w = tid >> 5;
  const int tM = blockIdx.y * 128, tN = blockIdx.x * 128;

  f32x8 acc[8];
#pragma unroll
  for (int nt = 0; nt < 8; ++nt) acc[nt] = f32x8{0,0,0,0,0,0,0,0};

#if USE_ASYNC
  const int nk = Kd / KSTEP;
  async_tile32(A,  lda, As[0], tM, 0, tid);
  async_tile32(Bt, ldb, Bs[0], tN, 0, tid);
  for (int kb = 0; kb < nk; ++kb) {
    const int cur = kb & 1, nxt = cur ^ 1;
    if (kb + 1 < nk) {
      async_tile32(A,  lda, As[nxt], tM, (kb + 1) * KSTEP, tid);
      async_tile32(Bt, ldb, Bs[nxt], tN, (kb + 1) * KSTEP, tid);
      WAIT_ASYNC(4);     // leave only the 4 just-issued outstanding
    } else {
      WAIT_ASYNC(0);
    }
    __syncthreads();     // current buffers visible to all waves
    bf16x16 af = frag_a<LSTR>(As[cur] + w * 16 * LSTR, lane);
#pragma unroll
    for (int nt = 0; nt < 8; ++nt) {
      bf16x16 bf = frag_b<LSTR>(Bs[cur] + nt * 16 * LSTR, lane);
      acc[nt] = __builtin_amdgcn_wmma_f32_16x16x32_bf16(
          false, af, false, bf, (short)0, acc[nt], false, false);
    }
    __syncthreads();     // all readers done before buffer is refilled
  }
#else
  for (int k0 = 0; k0 < Kd; k0 += KSTEP) {
    lds_load_tile(A,  lda, As[0], tM, k0, tid);
    lds_load_tile(Bt, ldb, Bs[0], tN, k0, tid);
    __syncthreads();
    bf16x16 af = frag_a<LSTR>(As[0] + w * 16 * LSTR, lane);
#pragma unroll
    for (int nt = 0; nt < 8; ++nt) {
      bf16x16 bf = frag_b<LSTR>(Bs[0] + nt * 16 * LSTR, lane);
      acc[nt] = __builtin_amdgcn_wmma_f32_16x16x32_bf16(
          false, af, false, bf, (short)0, acc[nt], false, false);
    }
    __syncthreads();
  }
#endif

  // C layout (ISA 7.12.2): VGPR i holds M = i + 8*(lane/16); N = lane%16.
  const int rbase = tM + w * 16 + ((lane >> 4) << 3);
  const int cIn = lane & 15;
#pragma unroll
  for (int nt = 0; nt < 8; ++nt) {
    int c = tN + nt * 16 + cIn;
    float bv = bias ? bias[c] : 0.f;
#pragma unroll
    for (int i = 0; i < 8; ++i) {
      float v = acc[nt][i] + bv;
      if (RELU) v = v > 0.f ? v : 0.f;
      size_t off = (size_t)(rbase + i) * ldc + c;
      if (OUT_BF16) ((__bf16*)Cout)[off] = (__bf16)v;
      else          ((float*)Cout)[off]  = v;
    }
  }
}

// ---------------------------------------------------------------------------
// Fused VQ: ids[n] = argmax_c ( Xn[n,:] . Cb[c,:] + bias[c] ), c < Kvalid.
// Async variant keeps the whole 128x256 X tile resident in LDS (66KB of the
// 320KB WGP LDS) and streams double-buffered codebook K-slices via ASYNCcnt.
// ---------------------------------------------------------------------------
__global__ __launch_bounds__(256)
void wmma_vq_argmax(const __bf16* __restrict__ Xn,   // [NROWS, EMB]
                    const __bf16* __restrict__ Cb,   // [Kp, EMB]
                    const float* __restrict__ bias,  // [Kp] or null
                    int Kp, int Kvalid, int* __restrict__ ids) {
  const int tid = threadIdx.x, lane = tid & 31, w = tid >> 5;
  const int tM = blockIdx.x * 128;
  const int cIn = lane & 15;

  float bestv[8]; int besti[8];
#pragma unroll
  for (int i = 0; i < 8; ++i) { bestv[i] = -3.4e38f; besti[i] = 0; }

#if USE_ASYNC
  __shared__ __bf16 Af[128 * AFSTR];       // full X tile, loaded once
  __shared__ __bf16 Bs[2][128 * LSTR];     // codebook K-slices, double buffered
  async_tile_fullA(Xn, Af, tM, tid);
  async_tile32(Cb, EMB, Bs[0], 0, 0, tid);
  const int nchunks = Kp / 128;
  int s = 0;
  for (int c0 = 0; c0 < nchunks; ++c0) {
    f32x8 acc[8];
#pragma unroll
    for (int nt = 0; nt < 8; ++nt) acc[nt] = f32x8{0,0,0,0,0,0,0,0};

    for (int kb = 0; kb < EMB / KSTEP; ++kb, ++s) {
      const int cur = s & 1, nxt = cur ^ 1;
      int nc = c0, nkb = kb + 1;
      if (nkb == EMB / KSTEP) { nc = c0 + 1; nkb = 0; }
      if (nc < nchunks) {
        async_tile32(Cb, EMB, Bs[nxt], nc * 128, nkb * KSTEP, tid);
        WAIT_ASYNC(2);
      } else {
        WAIT_ASYNC(0);
      }
      __syncthreads();
      bf16x16 af = frag_a<AFSTR>(Af + kb * KSTEP + w * 16 * AFSTR, lane);
#pragma unroll
      for (int nt = 0; nt < 8; ++nt) {
        bf16x16 bf = frag_b<LSTR>(Bs[cur] + nt * 16 * LSTR, lane);
        acc[nt] = __builtin_amdgcn_wmma_f32_16x16x32_bf16(
            false, af, false, bf, (short)0, acc[nt], false, false);
      }
      __syncthreads();
    }
#pragma unroll
    for (int nt = 0; nt < 8; ++nt) {
      int c = c0 * 128 + nt * 16 + cIn;
      float bv = bias ? bias[c] : 0.f;
      bool ok = c < Kvalid;
#pragma unroll
      for (int i = 0; i < 8; ++i) {
        float v = acc[nt][i] + bv;
        if (ok && v > bestv[i]) { bestv[i] = v; besti[i] = c; }
      }
    }
  }
#else
  __shared__ __bf16 As[128 * LSTR];
  __shared__ __bf16 Bs[128 * LSTR];
  for (int c0 = 0; c0 < Kp; c0 += 128) {
    f32x8 acc[8];
#pragma unroll
    for (int nt = 0; nt < 8; ++nt) acc[nt] = f32x8{0,0,0,0,0,0,0,0};
    for (int k0 = 0; k0 < EMB; k0 += KSTEP) {
      lds_load_tile(Xn, EMB, As, tM, k0, tid);
      lds_load_tile(Cb, EMB, Bs, c0, k0, tid);
      __syncthreads();
      bf16x16 af = frag_a<LSTR>(As + w * 16 * LSTR, lane);
#pragma unroll
      for (int nt = 0; nt < 8; ++nt) {
        bf16x16 bf = frag_b<LSTR>(Bs + nt * 16 * LSTR, lane);
        acc[nt] = __builtin_amdgcn_wmma_f32_16x16x32_bf16(
            false, af, false, bf, (short)0, acc[nt], false, false);
      }
      __syncthreads();
    }
#pragma unroll
    for (int nt = 0; nt < 8; ++nt) {
      int c = c0 + nt * 16 + cIn;
      float bv = bias ? bias[c] : 0.f;
      bool ok = c < Kvalid;
#pragma unroll
      for (int i = 0; i < 8; ++i) {
        float v = acc[nt][i] + bv;
        if (ok && v > bestv[i]) { bestv[i] = v; besti[i] = c; }
      }
    }
  }
#endif

  // reduce over the 16 lanes that share each row (xor masks stay in-group)
#pragma unroll
  for (int i = 0; i < 8; ++i) {
#pragma unroll
    for (int m = 1; m < 16; m <<= 1) {
      float ov = __shfl_xor(bestv[i], m, 32);
      int   oi = __shfl_xor(besti[i], m, 32);
      if (ov > bestv[i] || (ov == bestv[i] && oi < besti[i])) { bestv[i] = ov; besti[i] = oi; }
    }
  }
  if ((lane & 15) == 0) {
    int rbase = tM + w * 16 + ((lane >> 4) << 3);
#pragma unroll
    for (int i = 0; i < 8; ++i) ids[rbase + i] = besti[i];
  }
}

// ------------------------- small fp32 helper kernels -----------------------
__device__ __forceinline__ float block_sum256(float v, float* sm) {
#pragma unroll
  for (int m = 16; m >= 1; m >>= 1) v += __shfl_xor(v, m, 32);
  int lane = threadIdx.x & 31, w = threadIdx.x >> 5;
  if (lane == 0) sm[w] = v;
  __syncthreads();
  if (w == 0) {
    float t = (lane < 8) ? sm[lane] : 0.f;
#pragma unroll
    for (int m = 4; m >= 1; m >>= 1) t += __shfl_xor(t, m, 32);
    if (lane == 0) sm[0] = t;
  }
  __syncthreads();
  float s = sm[0];
  __syncthreads();
  return s;
}

__global__ void cvt_bf16(const float* __restrict__ s, __bf16* __restrict__ d, size_t n) {
  size_t i = (size_t)blockIdx.x * 256 + threadIdx.x;
  if (i < n) d[i] = (__bf16)s[i];
}

// W[K,Nf] fp32 -> Wt[Nf,K] bf16 (tiny one-time weight transpose)
__global__ void cvt_transpose_bf16(const float* __restrict__ W, __bf16* __restrict__ Wt,
                                   int K, int Nf) {
  size_t i = (size_t)blockIdx.x * 256 + threadIdx.x;
  if (i < (size_t)K * Nf) {
    int k = (int)(i / Nf), n = (int)(i % Nf);
    Wt[(size_t)n * K + k] = (__bf16)W[i];
  }
}

// row L2-normalize: fp32 [rows,256] -> bf16 (and optionally fp32). Pads with 0.
__global__ void rownorm256(const float* __restrict__ src, __bf16* __restrict__ dst_bf,
                           float* __restrict__ dst_f, int rows_valid) {
  __shared__ float sm[8];
  int r = blockIdx.x, t = threadIdx.x;
  float v = (r < rows_valid) ? src[(size_t)r * EMB + t] : 0.f;
  float s = block_sum256(v * v, sm);
  float nv = (r < rows_valid) ? v * rsqrtf(s + 1e-12f) : 0.f;
  dst_bf[(size_t)r * EMB + t] = (__bf16)nv;
  if (dst_f) dst_f[(size_t)r * EMB + t] = nv;
}

// Euclidean codebook prep: bf16 copy + bias[c] = -0.5*||c||^2
__global__ void euc_prep(const float* __restrict__ cb, __bf16* __restrict__ dst_bf,
                         float* __restrict__ biasv) {
  __shared__ float sm[8];
  int r = blockIdx.x, t = threadIdx.x;
  float v = cb[(size_t)r * EMB + t];
  float s = block_sum256(v * v, sm);
  dst_bf[(size_t)r * EMB + t] = (__bf16)v;
  if (t == 0) biasv[r] = -0.5f * s;
}

// cosine-VQ epilogue: loss += mean((q - l2norm(x))^2); r = x - q;
// optionally emit next-stage normalized input and accumulate qsum.
__global__ void update_cos(const float* __restrict__ xin, const int* __restrict__ ids,
                           const float* __restrict__ cbn, float* __restrict__ loss_acc,
                           float* __restrict__ r_out, __bf16* __restrict__ xnextn,
                           float* __restrict__ qsum) {
  __shared__ float sm[8];
  __shared__ int sidx;
  int n = blockIdx.x, t = threadIdx.x;
  if (t == 0) sidx = ids[n];
  __syncthreads();
  size_t off = (size_t)n * EMB + t;
  float xv = xin[off];
  float q  = cbn[(size_t)sidx * EMB + t];
  float s2 = block_sum256(xv * xv, sm);
  float xnv = xv * rsqrtf(s2 + 1e-12f);
  float d = q - xnv;
  float ls = block_sum256(d * d, sm);
  if (t == 0) atomicAdd(loss_acc, ls * (1.0f / ((float)NROWS * (float)EMB)));
  float r = xv - q;
  if (r_out)  r_out[off] = r;
  if (qsum)   qsum[off] += q;
  if (xnextn) {
    float s3 = block_sum256(r * r, sm);
    xnextn[off] = (__bf16)(r * rsqrtf(s3 + 1e-12f));
  }
}

// stage-3 input: x3 = res - (q1+q2), fp32 + raw bf16
__global__ void prep_euc_input(const float* __restrict__ res, const float* __restrict__ qsum,
                               float* __restrict__ x3, __bf16* __restrict__ x3b) {
  size_t i = (size_t)blockIdx.x * 256 + threadIdx.x;
  if (i < (size_t)NROWS * EMB) {
    float v = res[i] - qsum[i];
    x3[i] = v;
    x3b[i] = (__bf16)v;
  }
}

__global__ void update_euc(const float* __restrict__ x3, const int* __restrict__ ids,
                           const float* __restrict__ cb, float* __restrict__ loss_acc,
                           float* __restrict__ qsum) {
  __shared__ float sm[8];
  __shared__ int sidx;
  int n = blockIdx.x, t = threadIdx.x;
  if (t == 0) sidx = ids[n];
  __syncthreads();
  size_t off = (size_t)n * EMB + t;
  float q = cb[(size_t)sidx * EMB + t];
  float d = q - x3[off];
  float ls = block_sum256(d * d, sm);
  if (t == 0) atomicAdd(loss_acc, ls * (1.0f / ((float)NROWS * (float)EMB)));
  qsum[off] += q;
}

__global__ void zero_f32(float* __restrict__ p, size_t n) {
  size_t i = (size_t)blockIdx.x * 256 + threadIdx.x;
  if (i < n) p[i] = 0.f;
}

__global__ void write_ids(const int* __restrict__ i0, const int* __restrict__ i1,
                          const int* __restrict__ i2, const int* __restrict__ i3,
                          float* __restrict__ out) {
  size_t n = (size_t)blockIdx.x * 256 + threadIdx.x;
  if (n < NROWS) {
    out[n * 4 + 0] = (float)i0[n];
    out[n * 4 + 1] = (float)i1[n];
    out[n * 4 + 2] = (float)i2[n];
    out[n * 4 + 3] = (float)i3[n];
  }
}

__global__ void write_loss(const float* __restrict__ l, float* __restrict__ out) {
  if (threadIdx.x == 0 && blockIdx.x == 0)
    out[0] = 10.0f * (l[0] + l[1] + l[2] + l[3]);
}

// ---------------------------------------------------------------------------
extern "C" void kernel_launch(void* const* d_in, const int* in_sizes, int n_in,
                              void* d_out, int out_size, void* d_ws, size_t ws_size,
                              hipStream_t stream) {
  (void)in_sizes; (void)n_in; (void)out_size; (void)ws_size;
  const float* x       = (const float*)d_in[0];
  const float* enc_w1  = (const float*)d_in[1];
  const float* enc_b1  = (const float*)d_in[2];
  const float* enc_w2  = (const float*)d_in[3];
  const float* enc_b2  = (const float*)d_in[4];
  const float* cb_clus = (const float*)d_in[5];
  const float* cb_cos  = (const float*)d_in[6];   // [2, K, EMB]
  const float* cb_euc  = (const float*)d_in[7];
  float* out = (float*)d_out;

  // ---- workspace layout (bump; h/xb regions recycled) ----
  char* ws = (char*)d_ws;
  size_t o = 0;
  auto alloc = [&](size_t bytes) { size_t r = o; o += (bytes + 255) & ~(size_t)255; return r; };
  size_t o_xb    = alloc((size_t)NROWS * IN_DIM * 2);   // x bf16 (recycled as x3 f32 later)
  size_t o_w1t   = alloc((size_t)HID * IN_DIM * 2);
  size_t o_w2t   = alloc((size_t)EMB * HID * 2);
  size_t o_h     = alloc((size_t)NROWS * HID * 2);      // recycled: rr / rrn / x3b
  size_t o_res   = alloc((size_t)NROWS * EMB * 4);
  size_t o_resn  = alloc((size_t)NROWS * EMB * 2);
  size_t o_cbCb  = alloc((size_t)KCLUST_PAD * EMB * 2);
  size_t o_cbCf  = alloc((size_t)KCLUST_PAD * EMB * 4);
  size_t o_cb0b  = alloc((size_t)KCODES * EMB * 2);
  size_t o_cb0f  = alloc((size_t)KCODES * EMB * 4);
  size_t o_cb1b  = alloc((size_t)KCODES * EMB * 2);
  size_t o_cb1f  = alloc((size_t)KCODES * EMB * 4);
  size_t o_cbeb  = alloc((size_t)KCODES * EMB * 2);
  size_t o_biase = alloc((size_t)KCODES * 4);
  size_t o_ids   = alloc((size_t)4 * NROWS * 4);
  size_t o_loss  = alloc(4 * 4);

  __bf16* xb   = (__bf16*)(ws + o_xb);
  __bf16* w1t  = (__bf16*)(ws + o_w1t);
  __bf16* w2t  = (__bf16*)(ws + o_w2t);
  __bf16* h    = (__bf16*)(ws + o_h);
  float*  res  = (float*)(ws + o_res);
  __bf16* resn = (__bf16*)(ws + o_resn);
  __bf16* cbCb = (__bf16*)(ws + o_cbCb);
  float*  cbCf = (float*)(ws + o_cbCf);
  __bf16* cb0b = (__bf16*)(ws + o_cb0b);
  float*  cb0f = (float*)(ws + o_cb0f);
  __bf16* cb1b = (__bf16*)(ws + o_cb1b);
  float*  cb1f = (float*)(ws + o_cb1f);
  __bf16* cbeb = (__bf16*)(ws + o_cbeb);
  float*  biase= (float*)(ws + o_biase);
  int*    ids0 = (int*)(ws + o_ids);
  int*    ids1 = ids0 + NROWS;
  int*    ids2 = ids1 + NROWS;
  int*    ids3 = ids2 + NROWS;
  float*  loss = (float*)(ws + o_loss);
  // recycled regions (h is dead after GEMM2; xb dead after GEMM1)
  float*  rr   = (float*)(ws + o_h);                                  // [N,EMB] f32
  __bf16* rrn  = (__bf16*)(ws + o_h + (size_t)NROWS * EMB * 4);       // [N,EMB] bf16
  __bf16* x3b  = (__bf16*)(ws + o_h + (size_t)NROWS * EMB * 6);       // [N,EMB] bf16
  float*  x3   = (float*)(ws + o_xb);                                 // [N,EMB] f32
  float*  ids_out = out + 1;
  float*  qsum    = out + 1 + (size_t)NROWS * 4;                      // rq_embs region of d_out

  const dim3 blk(256);

  // 1) conversions / codebook prep
  cvt_bf16<<<dim3((NROWS * (size_t)IN_DIM + 255) / 256), blk, 0, stream>>>(x, xb, (size_t)NROWS * IN_DIM);
  cvt_transpose_bf16<<<dim3(((size_t)IN_DIM * HID + 255) / 256), blk, 0, stream>>>(enc_w1, w1t, IN_DIM, HID);
  cvt_transpose_bf16<<<dim3(((size_t)HID * EMB + 255) / 256), blk, 0, stream>>>(enc_w2, w2t, HID, EMB);
  rownorm256<<<dim3(KCLUST_PAD), blk, 0, stream>>>(cb_clus, cbCb, cbCf, KCLUST);
  rownorm256<<<dim3(KCODES), blk, 0, stream>>>(cb_cos, cb0b, cb0f, KCODES);
  rownorm256<<<dim3(KCODES), blk, 0, stream>>>(cb_cos + (size_t)KCODES * EMB, cb1b, cb1f, KCODES);
  euc_prep<<<dim3(KCODES), blk, 0, stream>>>(cb_euc, cbeb, biase);

  // 2) encoder MLP on WMMA
  wmma_gemm_nt<1, 1><<<dim3(HID / 128, NROWS / 128), blk, 0, stream>>>(
      xb, IN_DIM, w1t, IN_DIM, enc_b1, (void*)h, HID, IN_DIM);
  wmma_gemm_nt<0, 0><<<dim3(EMB / 128, NROWS / 128), blk, 0, stream>>>(
      h, HID, w2t, HID, enc_b2, (void*)res, EMB, HID);

  // 3) res row-norm + accumulator init
  rownorm256<<<dim3(NROWS), blk, 0, stream>>>(res, resn, nullptr, NROWS);
  zero_f32<<<dim3(1), blk, 0, stream>>>(loss, 4);
  zero_f32<<<dim3(((size_t)NROWS * EMB + 255) / 256), blk, 0, stream>>>(qsum, (size_t)NROWS * EMB);

  // 4) stage 1: cluster cosine VQ (50 codes, padded to 128)
  wmma_vq_argmax<<<dim3(NROWS / 128), blk, 0, stream>>>(resn, cbCb, nullptr, KCLUST_PAD, KCLUST, ids0);
  update_cos<<<dim3(NROWS), blk, 0, stream>>>(res, ids0, cbCf, loss + 0, rr, rrn, nullptr);

  // 5) stage 2: residual cosine VQ, 2 quantizers (qsum accumulates q1+q2)
  wmma_vq_argmax<<<dim3(NROWS / 128), blk, 0, stream>>>(rrn, cb0b, nullptr, KCODES, KCODES, ids1);
  update_cos<<<dim3(NROWS), blk, 0, stream>>>(rr, ids1, cb0f, loss + 1, rr, rrn, qsum);
  wmma_vq_argmax<<<dim3(NROWS / 128), blk, 0, stream>>>(rrn, cb1b, nullptr, KCODES, KCODES, ids2);
  update_cos<<<dim3(NROWS), blk, 0, stream>>>(rr, ids2, cb1f, loss + 2, nullptr, nullptr, qsum);

  // 6) stage 3: Euclidean VQ on res - (q1+q2)  [rq_embs overwrite quirk]
  prep_euc_input<<<dim3(((size_t)NROWS * EMB + 255) / 256), blk, 0, stream>>>(res, qsum, x3, x3b);
  wmma_vq_argmax<<<dim3(NROWS / 128), blk, 0, stream>>>(x3b, cbeb, biase, KCODES, KCODES, ids3);
  update_euc<<<dim3(NROWS), blk, 0, stream>>>(x3, ids3, cb_euc, loss + 3, qsum);

  // 7) outputs: [total_loss | sem_ids (as float) | rq_embs]
  write_ids<<<dim3((NROWS + 255) / 256), blk, 0, stream>>>(ids0, ids1, ids2, ids3, ids_out);
  write_loss<<<dim3(1), dim3(32), 0, stream>>>(loss, out);
}
